// Informer_33432025432553
// MI455X (gfx1250) — matrix operand order repaired and checked by reading, not compiled, not run on hardware
//
#include <hip/hip_runtime.h>
#include <hip/hip_bf16.h>
#include <cmath>

// ---------------------------------------------------------------------------
// Types for CDNA5 WMMA
// ---------------------------------------------------------------------------
typedef __attribute__((ext_vector_type(16))) __bf16 v16bf;
typedef __attribute__((ext_vector_type(8)))  __bf16 v8bf;
typedef __attribute__((ext_vector_type(4)))  __bf16 v4bf;
typedef __attribute__((ext_vector_type(8)))  float  v8f;

union bfrag { v8bf h[2]; v16bf v; };

// LDS row stride: 40 bf16 = 80 B (keeps 16-B aligned chunks, staggers banks)
#define LSTR 40

// ---------------------------------------------------------------------------
// WMMA inner step on one 64(M)x64(N)x32(K) tile.
// LDS layouts chosen so each fragment = two 16-byte ds loads:
//   As[m][k]        : A fragment lane needs k = 8*khalf..+7 and 16+8*khalf..+7
//   Bs[n][k] (=W!)  : B fragment lane needs k = 8*khalf..+7 and 16+8*khalf..+7 of row ncol
// (layouts per CDNA5 ISA 7.12.2, wave32)
// ---------------------------------------------------------------------------
__device__ __forceinline__ void wmma_tile(const __bf16 (&As)[64][LSTR],
                                          const __bf16 (&Bs)[64][LSTR],
                                          int wave, int lane, v8f* acc) {
  const int khalf = lane >> 4;
  const int mrow  = wave * 16 + (lane & 15);
  bfrag a;
  a.h[0] = *(const v8bf*)&As[mrow][8 * khalf];
  a.h[1] = *(const v8bf*)&As[mrow][16 + 8 * khalf];
  const int kbase = 8 * khalf;
#pragma unroll
  for (int ns = 0; ns < 4; ++ns) {
    const int ncol = ns * 16 + (lane & 15);
    bfrag b;
    b.h[0] = *(const v8bf*)&Bs[ncol][kbase];
    b.h[1] = *(const v8bf*)&Bs[ncol][kbase + 16];
    acc[ns] = __builtin_amdgcn_wmma_f32_16x16x32_bf16(
        false, a.v, false, b.v, (short)0, acc[ns], false, false);
  }
}

// Stage a 64-row x 32-col fp32 tile into bf16 LDS (native cvt, 8B ds stores).
__device__ __forceinline__ void stage_A(__bf16 (&S)[64][LSTR],
                                        const float* __restrict__ src,
                                        int row0, int ld, int kb, int tid) {
  const int r0 = tid >> 3, kc = (tid & 7) * 4;
#pragma unroll
  for (int rr = 0; rr < 64; rr += 16) {
    const int r = r0 + rr;
    const float4 f = *(const float4*)(src + (size_t)(row0 + r) * ld + kb + kc);
    v4bf o;
    o[0] = (__bf16)f.x; o[1] = (__bf16)f.y; o[2] = (__bf16)f.z; o[3] = (__bf16)f.w;
    *(v4bf*)&S[r][kc] = o;
  }
}

// Stage a 64x32 bf16 weight tile with the CDNA5 async VMEM->LDS path.
// 256 16-byte lane-transfers (2 per thread), no VGPR round-trip, ASYNCcnt.
__device__ __forceinline__ void stage_B_async(__bf16 (&S)[64][LSTR],
                                              const __bf16* __restrict__ Wbf,
                                              int n0, int K, int kb, int tid) {
#pragma unroll
  for (int it = 0; it < 2; ++it) {
    const int id = tid + it * 128;          // 0..255
    const int r = id >> 2, c8 = (id & 3) * 8;
    unsigned lds  = (unsigned)(size_t)&S[r][c8];   // LDS offset = addr[31:0]
    unsigned goff = ((unsigned)(n0 + r) * (unsigned)K + (unsigned)(kb + c8)) * 2u;
    asm volatile("global_load_async_to_lds_b128 %0, %1, %2 offset:0"
                 :: "v"(lds), "v"(goff), "s"(Wbf) : "memory");
  }
}

// ---------------------------------------------------------------------------
// GEMM: C[M,N] = A[M,K] @ W[N,K]^T + bias (+ residual) (+ exact GELU)
// fp32 activations -> bf16 (cvt in A staging); weights pre-converted bf16.
// Double-buffered LDS; async B staging fenced with s_wait_asynccnt.
// M%64==0, N%64==0, K%32==0.
// ---------------------------------------------------------------------------
template <int ACT, bool HAS_RES>
__global__ __launch_bounds__(128) void gemm_bf16_wmma(
    const float* __restrict__ A, const __bf16* __restrict__ W,
    const float* __restrict__ bias, const float* __restrict__ res,
    float* __restrict__ C, int M, int N, int K) {
  __shared__ __bf16 As[2][64][LSTR];
  __shared__ __bf16 Bs[2][64][LSTR];
  const int tid = threadIdx.x, wave = tid >> 5, lane = tid & 31;
  const int m0 = blockIdx.y * 64, n0 = blockIdx.x * 64;
  v8f acc[4] = {};
  const int nk = K >> 5;
  stage_A(As[0], A, m0, K, 0, tid);
  stage_B_async(Bs[0], W, n0, K, 0, tid);
  for (int ki = 0; ki < nk; ++ki) {
    asm volatile("s_wait_asynccnt 0" ::: "memory");  // async B tile landed
    __syncthreads();                                 // + ds stores visible
    if (ki + 1 < nk) {                               // prefetch next K-step
      stage_A(As[(ki + 1) & 1], A, m0, K, (ki + 1) << 5, tid);
      stage_B_async(Bs[(ki + 1) & 1], W, n0, K, (ki + 1) << 5, tid);
    }
    wmma_tile(As[ki & 1], Bs[ki & 1], wave, lane, acc);
  }
#pragma unroll
  for (int ns = 0; ns < 4; ++ns)
#pragma unroll
    for (int r = 0; r < 8; ++r) {
      int m = m0 + wave * 16 + r + 8 * (lane >> 4);
      int n = n0 + ns * 16 + (lane & 15);
      float v = acc[ns][r] + bias[n];
      if (HAS_RES) v += res[(size_t)m * N + n];
      if (ACT == 1) v = 0.5f * v * (1.0f + erff(v * 0.70710678118654752440f));
      C[(size_t)m * N + n] = v;
    }
}

// ---------------------------------------------------------------------------
// Circular conv1d (kernel 3, wrap pad) as WMMA GEMM, K folded = 3*512.
// 48 pipelined steps: step s -> tap t3 = s/16, k-block kb = (s%16)*32.
// Weights pre-converted bf16 (stride-3 gather is now a bit-copy).
// ---------------------------------------------------------------------------
__device__ __forceinline__ void conv_stage(__bf16 (&SA)[64][LSTR],
                                           __bf16 (&SB)[64][LSTR],
                                           const float* __restrict__ X,
                                           const __bf16* __restrict__ Wc,
                                           int m0, int n0, int L, int s, int tid) {
  const int t3 = s >> 4, kb = (s & 15) << 5;
  const int r0 = tid >> 3, kc = (tid & 7) * 4;
#pragma unroll
  for (int rr = 0; rr < 64; rr += 16) {
    const int r = r0 + rr, m = m0 + r, bb = m / L, l = m % L;
    int lw = l - 1 + t3;
    lw += (lw < 0) ? L : 0;
    lw -= (lw >= L) ? L : 0;
    const float4 f = *(const float4*)(X + ((size_t)(bb * L + lw) * 512) + kb + kc);
    v4bf o;
    o[0] = (__bf16)f.x; o[1] = (__bf16)f.y; o[2] = (__bf16)f.z; o[3] = (__bf16)f.w;
    *(v4bf*)&SA[r][kc] = o;
  }
  for (int i = tid; i < 64 * 32; i += 128) {
    const int n = i >> 5, k = i & 31;
    SB[n][k] = Wc[(((size_t)(n0 + n) * 512) + (kb + k)) * 3 + t3];
  }
}

__global__ __launch_bounds__(128) void conv_wmma(
    const float* __restrict__ X, const __bf16* __restrict__ Wc,
    const float* __restrict__ bias, float* __restrict__ H, int L) {
  __shared__ __bf16 As[2][64][LSTR];
  __shared__ __bf16 Bs[2][64][LSTR];
  const int tid = threadIdx.x, wave = tid >> 5, lane = tid & 31;
  const int m0 = blockIdx.y * 64, n0 = blockIdx.x * 64;
  v8f acc[4] = {};
  conv_stage(As[0], Bs[0], X, Wc, m0, n0, L, 0, tid);
  for (int s = 0; s < 48; ++s) {
    __syncthreads();
    if (s + 1 < 48)
      conv_stage(As[(s + 1) & 1], Bs[(s + 1) & 1], X, Wc, m0, n0, L, s + 1, tid);
    wmma_tile(As[s & 1], Bs[s & 1], wave, lane, acc);
  }
#pragma unroll
  for (int ns = 0; ns < 4; ++ns)
#pragma unroll
    for (int r = 0; r < 8; ++r) {
      int m = m0 + wave * 16 + r + 8 * (lane >> 4);
      int n = n0 + ns * 16 + (lane & 15);
      H[(size_t)m * 512 + n] = acc[ns][r] + bias[n];
    }
}

// ---------------------------------------------------------------------------
// fp32 -> bf16 weight pre-conversion (once per launch, amortized over all
// 288x8 tile re-stages that previously each re-converted the same weights).
// ---------------------------------------------------------------------------
__global__ void cvt_bf16(const float* __restrict__ X, __bf16* __restrict__ O,
                         int n) {
  int i = blockIdx.x * blockDim.x + threadIdx.x;
  if (i < n) O[i] = (__bf16)X[i];
}

// ---------------------------------------------------------------------------
// Full attention, one block per (b,h).  L<=36, D=64.  Tiny vs GEMMs -> VALU.
// ---------------------------------------------------------------------------
template <bool CAUSAL>
__global__ __launch_bounds__(64) void attn_full(
    const float* __restrict__ Q, const float* __restrict__ K,
    const float* __restrict__ V, float* __restrict__ O, int Lq, int Lk) {
  const int bh = blockIdx.x, b = bh >> 3, h = bh & 7;
  const int l = threadIdx.x;
  if (l >= Lq) return;
  const float* qp = Q + ((size_t)(b * Lq + l) * 512) + h * 64;
  float sc[40];
  float mx = -__builtin_inff();
  for (int k = 0; k < Lk; ++k) {
    float s = -__builtin_inff();
    if (!(CAUSAL && k > l)) {
      const float* kp = K + ((size_t)(b * Lk + k) * 512) + h * 64;
      s = 0.f;
      for (int d = 0; d < 64; ++d) s += qp[d] * kp[d];
      s *= 0.125f;  // 1/sqrt(64)
    }
    sc[k] = s;
    mx = fmaxf(mx, s);
  }
  float Z = 0.f;
  for (int k = 0; k < Lk; ++k) { sc[k] = expf(sc[k] - mx); Z += sc[k]; }
  float invZ = 1.0f / Z;
  float* op = O + ((size_t)(b * Lq + l) * 512) + h * 64;
  for (int d = 0; d < 64; ++d) {
    float s = 0.f;
    for (int k = 0; k < Lk; ++k)
      s += sc[k] * V[((size_t)(b * Lk + k) * 512) + h * 64 + d];
    op[d] = s * invZ;
  }
}

// ---------------------------------------------------------------------------
// ProbSparse attention (encoder).  Device hash RNG stands in for the host
// numpy sample stream; structure preserved: M = max(QK_s) - sum(QK_s)/Lk,
// top-u queries -> softmax attention, rest -> mean(V).
// ---------------------------------------------------------------------------
__global__ __launch_bounds__(64) void attn_prob(
    const float* __restrict__ Q, const float* __restrict__ K,
    const float* __restrict__ V, float* __restrict__ O, int L, int Up, int u) {
  __shared__ float Ms[64];
  __shared__ int   sel[64];
  const int bh = blockIdx.x, b = bh >> 3, h = bh & 7;
  const int l = threadIdx.x;
  sel[l] = 0;
  if (l < L) {
    const float* qp = Q + ((size_t)(b * L + l) * 512) + h * 64;
    float mval = -__builtin_inff(), ssum = 0.f;
    for (int j = 0; j < Up; ++j) {
      unsigned idx = (2654435761u * (unsigned)(l * 131 + j) + 0x9E3779B9u) % (unsigned)L;
      const float* kp = K + ((size_t)(b * L + idx) * 512) + h * 64;
      float s = 0.f;
      for (int d = 0; d < 64; ++d) s += qp[d] * kp[d];
      mval = fmaxf(mval, s);
      ssum += s;
    }
    Ms[l] = mval - ssum / (float)L;
  }
  __syncthreads();
  if (threadIdx.x == 0) {  // top-u selection (L<=36, ties -> lowest index)
    for (int t = 0; t < u; ++t) {
      float best = -__builtin_inff(); int bi = 0;
      for (int q = 0; q < L; ++q)
        if (!sel[q] && Ms[q] > best) { best = Ms[q]; bi = q; }
      sel[bi] = 1;
    }
  }
  __syncthreads();
  if (l >= L) return;
  float* op = O + ((size_t)(b * L + l) * 512) + h * 64;
  if (sel[l]) {
    const float* qp = Q + ((size_t)(b * L + l) * 512) + h * 64;
    float sc[40];
    float mx = -__builtin_inff();
    for (int k = 0; k < L; ++k) {
      const float* kp = K + ((size_t)(b * L + k) * 512) + h * 64;
      float s = 0.f;
      for (int d = 0; d < 64; ++d) s += qp[d] * kp[d];
      s *= 0.125f;
      sc[k] = s;
      mx = fmaxf(mx, s);
    }
    float Z = 0.f;
    for (int k = 0; k < L; ++k) { sc[k] = expf(sc[k] - mx); Z += sc[k]; }
    float invZ = 1.0f / Z;
    for (int d = 0; d < 64; ++d) {
      float s = 0.f;
      for (int k = 0; k < L; ++k)
        s += sc[k] * V[((size_t)(b * L + k) * 512) + h * 64 + d];
      op[d] = s * invZ;
    }
  } else {
    float invL = 1.0f / (float)L;
    for (int d = 0; d < 64; ++d) {
      float s = 0.f;
      for (int k = 0; k < L; ++k)
        s += V[((size_t)(b * L + k) * 512) + h * 64 + d];
      op[d] = s * invL;
    }
  }
}

// ---------------------------------------------------------------------------
// Fused (optional residual add +) LayerNorm over last dim (512).
// ---------------------------------------------------------------------------
__global__ __launch_bounds__(128) void add_ln(
    const float* __restrict__ A, const float* __restrict__ R,
    const float* __restrict__ g, const float* __restrict__ be,
    float* __restrict__ O) {
  __shared__ float red[128];
  const int m = blockIdx.x, t = threadIdx.x;
  const float* a = A + (size_t)m * 512;
  float v[4]; float s = 0.f;
#pragma unroll
  for (int i = 0; i < 4; ++i) {
    float x = a[t + 128 * i];
    if (R) x += R[(size_t)m * 512 + t + 128 * i];
    v[i] = x; s += x;
  }
  red[t] = s; __syncthreads();
  for (int o = 64; o > 0; o >>= 1) { if (t < o) red[t] += red[t + o]; __syncthreads(); }
  float mu = red[0] * (1.0f / 512.0f);
  __syncthreads();
  float q = 0.f;
#pragma unroll
  for (int i = 0; i < 4; ++i) { float d = v[i] - mu; q += d * d; }
  red[t] = q; __syncthreads();
  for (int o = 64; o > 0; o >>= 1) { if (t < o) red[t] += red[t + o]; __syncthreads(); }
  float inv = rsqrtf(red[0] * (1.0f / 512.0f) + 1e-5f);
#pragma unroll
  for (int i = 0; i < 4; ++i) {
    int c = t + 128 * i;
    O[(size_t)m * 512 + c] = (v[i] - mu) * inv * g[c] + be[c];
  }
}

// ---------------------------------------------------------------------------
// BatchNorm stats per channel, then BN + ELU + maxpool(3, stride 2, -inf pad).
// ---------------------------------------------------------------------------
__global__ __launch_bounds__(256) void bn_stats(
    const float* __restrict__ H, float* __restrict__ mean,
    float* __restrict__ var, int M) {
  __shared__ float s1[256], s2[256];
  const int c = blockIdx.x, t = threadIdx.x;
  float a = 0.f, b = 0.f;
  for (int m = t; m < M; m += 256) {
    float x = H[(size_t)m * 512 + c];
    a += x; b += x * x;
  }
  s1[t] = a; s2[t] = b; __syncthreads();
  for (int o = 128; o > 0; o >>= 1) {
    if (t < o) { s1[t] += s1[t + o]; s2[t] += s2[t + o]; }
    __syncthreads();
  }
  if (t == 0) {
    float mu = s1[0] / (float)M;
    mean[c] = mu;
    var[c] = s2[0] / (float)M - mu * mu;
  }
}

__global__ void bn_elu_pool(const float* __restrict__ H,
                            const float* __restrict__ mean,
                            const float* __restrict__ var,
                            const float* __restrict__ bng,
                            const float* __restrict__ bnb,
                            float* __restrict__ O, int Bn, int L, int Lout) {
  int idx = blockIdx.x * blockDim.x + threadIdx.x;
  int total = Bn * Lout * 512;
  if (idx >= total) return;
  int c = idx & 511, rest = idx >> 9;
  int lo = rest % Lout, b = rest / Lout;
  float mu = mean[c];
  float inv = rsqrtf(var[c] + 1e-5f);
  float gg = bng[c], bb = bnb[c];
  float best = -__builtin_inff();
  for (int w = 0; w < 3; ++w) {
    int l = 2 * lo - 1 + w;
    if (l < 0 || l >= L) continue;
    float x = H[((size_t)(b * L + l) * 512) + c];
    x = (x - mu) * inv * gg + bb;
    x = (x > 0.f) ? x : (expf(x) - 1.0f);  // ELU
    best = fmaxf(best, x);
  }
  O[((size_t)(b * Lout + lo) * 512) + c] = best;
}

// ---------------------------------------------------------------------------
// Sinusoidal positional embedding add (X==nullptr -> pure PE broadcast).
// ---------------------------------------------------------------------------
__global__ void pos_add(const float* __restrict__ X, float* __restrict__ O,
                        int M, int L) {
  int idx = blockIdx.x * blockDim.x + threadIdx.x;
  if (idx >= M * 512) return;
  int c = idx & 511, m = idx >> 9, l = m % L;
  int p = c >> 1;
  float div = expf((float)(2 * p) * (-9.210340371976184f / 512.0f));
  float ang = (float)l * div;
  float pe = (c & 1) ? cosf(ang) : sinf(ang);
  O[idx] = (X ? X[idx] : 0.f) + pe;
}

// ---------------------------------------------------------------------------
// Final projection 512 -> 1.
// ---------------------------------------------------------------------------
__global__ __launch_bounds__(128) void proj_kernel(
    const float* __restrict__ X, const float* __restrict__ w,
    const float* __restrict__ b, float* __restrict__ out, int M) {
  __shared__ float red[128];
  const int m = blockIdx.x, t = threadIdx.x;
  float s = 0.f;
#pragma unroll
  for (int i = 0; i < 4; ++i) {
    int c = t + 128 * i;
    s += X[(size_t)m * 512 + c] * w[c];
  }
  red[t] = s; __syncthreads();
  for (int o = 64; o > 0; o >>= 1) { if (t < o) red[t] += red[t + o]; __syncthreads(); }
  if (t == 0) out[m] = red[0] + b[0];
}

// ---------------------------------------------------------------------------
// Host-side GEMM dispatcher (W = pre-converted bf16)
// ---------------------------------------------------------------------------
static void gemm(hipStream_t s, const float* A, const __bf16* W, const float* bias,
                 const float* res, float* C, int M, int N, int K, int act) {
  dim3 grid(N / 64, M / 64), blk(128);
  if (res) {
    if (act) gemm_bf16_wmma<1, true><<<grid, blk, 0, s>>>(A, W, bias, res, C, M, N, K);
    else     gemm_bf16_wmma<0, true><<<grid, blk, 0, s>>>(A, W, bias, res, C, M, N, K);
  } else {
    if (act) gemm_bf16_wmma<1, false><<<grid, blk, 0, s>>>(A, W, bias, res, C, M, N, K);
    else     gemm_bf16_wmma<0, false><<<grid, blk, 0, s>>>(A, W, bias, res, C, M, N, K);
  }
}

// ---------------------------------------------------------------------------
// Param flattening (jax pytree: dicts in sorted-key order):
//  0: x_enc
//  conv[i] (i=0,1) at 1+4i:   b, bn_b, bn_g, w
//  dec[i]  (i=0,1) at 9+26i:  conv1{b,w} conv2{b,w} cross{k.b,k.w,o.b,o.w,
//           q.b,q.w,v.b,v.w} norm1{b,g} norm2{b,g} norm3{b,g}
//           self{k.b,k.w,o.b,o.w,q.b,q.w,v.b,v.w}
//  dec_norm: 61 b, 62 g
//  enc[i] (i=0..2) at 63+16i: attn{k.b,k.w,o.b,o.w,q.b,q.w,v.b,v.w}
//           conv1{b,w} conv2{b,w} norm1{b,g} norm2{b,g}
//  enc_norm: 111 b, 112 g ;  proj: 113 b, 114 w
// ---------------------------------------------------------------------------
extern "C" void kernel_launch(void* const* d_in, const int* in_sizes, int n_in,
                              void* d_out, int out_size, void* d_ws, size_t ws_size,
                              hipStream_t stream) {
  auto in = [&](int i) { return (const float*)d_in[i]; };

  float* ws = (float*)d_ws;
  const size_t BUF = (size_t)512 * 36 * 512;  // largest token-buffer (37.75 MB)
  float* bX  = ws + 0 * BUF;
  float* bQ  = ws + 1 * BUF;
  float* bK  = ws + 2 * BUF;
  float* bV  = ws + 3 * BUF;
  float* bA  = ws + 4 * BUF;
  float* bT1 = ws + 5 * BUF;
  float* bT2 = ws + 6 * BUF;
  float* bE  = ws + 7 * BUF;
  float* bMean = ws + 8 * BUF;
  float* bVar  = bMean + 512;

  // bf16 weight pool (converted once per launch, 16-B aligned)
  __bf16* wpool = (__bf16*)(ws + 8 * BUF + 1024);
  size_t woff = 0;
  auto cvt = [&](int idx, size_t n) -> const __bf16* {
    __bf16* dst = wpool + woff;
    cvt_bf16<<<((int)n + 255) / 256, 256, 0, stream>>>(in(idx), dst, (int)n);
    woff += n;
    return dst;
  };
  const size_t WSZ = 512 * 512, CSZ = 512 * 512 * 3;

  const int B = 512;

  // ---- encoder embedding ----
  {
    int M = B * 36, tot = M * 512;
    pos_add<<<(tot + 255) / 256, 256, 0, stream>>>(in(0), bX, M, 36);
  }

  const int Ls[3] = {36, 18, 9};
  for (int li = 0; li < 3; ++li) {
    const int L = Ls[li], M = B * L;
    const int base = 63 + 16 * li;
    const float *kb_ = in(base + 0), *ob = in(base + 2), *qb = in(base + 4),
                *vb = in(base + 6);
    const __bf16 *kw = cvt(base + 1, WSZ), *ow = cvt(base + 3, WSZ),
                 *qw = cvt(base + 5, WSZ), *vw = cvt(base + 7, WSZ);
    const float *c1b = in(base + 8), *c2b = in(base + 10);
    const __bf16 *c1w = cvt(base + 9, WSZ), *c2w = cvt(base + 11, WSZ);
    const float *n1b = in(base + 12), *n1g = in(base + 13),
                *n2b = in(base + 14), *n2g = in(base + 15);

    gemm(stream, bX, qw, qb, nullptr, bQ, M, 512, 512, 0);
    gemm(stream, bX, kw, kb_, nullptr, bK, M, 512, 512, 0);
    gemm(stream, bX, vw, vb, nullptr, bV, M, 512, 512, 0);

    int cl = (int)ceilf(logf((float)L));
    int Up = 5 * cl; if (Up > L) Up = L;
    int u  = 5 * cl; if (u > L)  u = L;
    attn_prob<<<B * 8, 64, 0, stream>>>(bQ, bK, bV, bA, L, Up, u);

    gemm(stream, bA, ow, ob, bX, bT1, M, 512, 512, 0);           // x + attn
    add_ln<<<M, 128, 0, stream>>>(bT1, nullptr, n1g, n1b, bT2);  // norm1
    gemm(stream, bT2, c1w, c1b, nullptr, bT1, M, 512, 512, 1);   // gelu(conv1)
    gemm(stream, bT1, c2w, c2b, bT2, bQ, M, 512, 512, 0);        // x + conv2
    add_ln<<<M, 128, 0, stream>>>(bQ, nullptr, n2g, n2b, bX);    // norm2

    if (li < 2) {  // distil
      const int cb = 1 + 4 * li;
      const float *cbias = in(cb + 0), *bnb = in(cb + 1), *bng = in(cb + 2);
      const __bf16* cw = cvt(cb + 3, CSZ);
      dim3 g(512 / 64, M / 64);
      conv_wmma<<<g, 128, 0, stream>>>(bX, cw, cbias, bT1, L);
      bn_stats<<<512, 256, 0, stream>>>(bT1, bMean, bVar, M);
      int Lout = (L - 1) / 2 + 1;
      int tot = B * Lout * 512;
      bn_elu_pool<<<(tot + 255) / 256, 256, 0, stream>>>(
          bT1, bMean, bVar, bng, bnb, bX, B, L, Lout);
    }
  }

  // ---- encoder final norm ----
  const int Me = B * 9;
  add_ln<<<Me, 128, 0, stream>>>(bX, nullptr, in(112), in(111), bE);

  // ---- decoder ----
  const int Md = B * 24;
  {
    int tot = Md * 512;
    pos_add<<<(tot + 255) / 256, 256, 0, stream>>>(nullptr, bX, Md, 24);
  }
  for (int li = 0; li < 2; ++li) {
    const int base = 9 + 26 * li;
    const float *c1b = in(base + 0), *c2b = in(base + 2);
    const __bf16 *c1w = cvt(base + 1, WSZ), *c2w = cvt(base + 3, WSZ);
    const float *xkb = in(base + 4), *xob = in(base + 6), *xqb = in(base + 8),
                *xvb = in(base + 10);
    const __bf16 *xkw = cvt(base + 5, WSZ), *xow = cvt(base + 7, WSZ),
                 *xqw = cvt(base + 9, WSZ), *xvw = cvt(base + 11, WSZ);
    const float *n1b = in(base + 12), *n1g = in(base + 13),
                *n2b = in(base + 14), *n2g = in(base + 15),
                *n3b = in(base + 16), *n3g = in(base + 17);
    const float *skb = in(base + 18), *sob = in(base + 20), *sqb = in(base + 22),
                *svb = in(base + 24);
    const __bf16 *skw = cvt(base + 19, WSZ), *sow = cvt(base + 21, WSZ),
                 *sqw = cvt(base + 23, WSZ), *svw = cvt(base + 25, WSZ);

    // self attention (causal)
    gemm(stream, bX, sqw, sqb, nullptr, bQ, Md, 512, 512, 0);
    gemm(stream, bX, skw, skb, nullptr, bK, Md, 512, 512, 0);
    gemm(stream, bX, svw, svb, nullptr, bV, Md, 512, 512, 0);
    attn_full<true><<<B * 8, 64, 0, stream>>>(bQ, bK, bV, bA, 24, 24);
    gemm(stream, bA, sow, sob, bX, bT1, Md, 512, 512, 0);
    add_ln<<<Md, 128, 0, stream>>>(bT1, nullptr, n1g, n1b, bX);

    // cross attention
    gemm(stream, bX, xqw, xqb, nullptr, bQ, Md, 512, 512, 0);
    gemm(stream, bE, xkw, xkb, nullptr, bK, Me, 512, 512, 0);
    gemm(stream, bE, xvw, xvb, nullptr, bV, Me, 512, 512, 0);
    attn_full<false><<<B * 8, 64, 0, stream>>>(bQ, bK, bV, bA, 24, 9);
    gemm(stream, bA, xow, xob, bX, bT1, Md, 512, 512, 0);
    add_ln<<<Md, 128, 0, stream>>>(bT1, nullptr, n2g, n2b, bX);

    // FFN
    gemm(stream, bX, c1w, c1b, nullptr, bT1, Md, 512, 512, 1);
    gemm(stream, bT1, c2w, c2b, bX, bT2, Md, 512, 512, 0);
    add_ln<<<Md, 128, 0, stream>>>(bT2, nullptr, n3g, n3b, bX);
  }

  // ---- final norm + projection ----
  add_ln<<<Md, 128, 0, stream>>>(bX, nullptr, in(62), in(61), bT1);
  proj_kernel<<<Md, 128, 0, stream>>>(bT1, in(114), in(113), (float*)d_out, Md);
}
